// KeywordSpottingModel_29377576305206
// MI455X (gfx1250) — compile-verified, matrix operand order checked by reading, not run on hardware
//
#include <hip/hip_runtime.h>

typedef __attribute__((ext_vector_type(16))) _Float16 v16h;
typedef __attribute__((ext_vector_type(8)))  _Float16 v8h;
typedef __attribute__((ext_vector_type(8)))  float    v8f;

#define B_    16
#define IN_   80
#define L_    512
#define DM_   512
#define DS_   16
#define DC_   4
#define DI_   1024
#define DTR_  32
#define NL_   2
#define NLAB_ 35
#define ML_   (B_ * L_)   /* 8192 GEMM rows */
#define INP_  96          /* IN padded to multiple of 32 */
#define NFCP_ 48          /* NLAB padded to multiple of 16 */

// ---------------------------------------------------------------------------
// WMMA fragment layouts per CDNA5 ISA 7.12.2 (16-bit, 16x16x32):
//   A (16x32): lane m = lane&15, K halves [k0..k0+7],[k0+16..k0+23], k0=(lane>>4)*8
//              -> two contiguous v8h (b128) loads per fragment
//   B (32x16): same pattern, n = lane&15 over W rows (W stored N x K row-major)
//   C/D (16x16 f32): lane n = lane&15, VGPR r holds M = r + (lane>=16 ? 8 : 0)
// EPI: 0 = +bias (optional), 1 = softplus(+bias), 2 = x*scale[n]+shift[n] (BN)
// ---------------------------------------------------------------------------

__device__ __forceinline__ v16h frag_combine(v8h lo, v8h hi)
{
    v16h f;
#pragma unroll
    for (int i = 0; i < 8; ++i) { f[i] = lo[i]; f[i + 8] = hi[i]; }
    return f;
}

template<int EPI>
__device__ __forceinline__ void epi_store(
    float* __restrict__ C, int ldc, int mbase, int half, int n, int Nbound,
    v8f acc, const float* __restrict__ bias,
    const float* __restrict__ scale, const float* __restrict__ shift)
{
    if (n >= Nbound) return;
    float bv = (EPI != 2 && bias) ? bias[n] : 0.f;
    float sc = (EPI == 2) ? scale[n] : 1.f;
    float sh = (EPI == 2) ? shift[n] : 0.f;
#pragma unroll
    for (int r = 0; r < 8; ++r) {
        int m = mbase + half * 8 + r;
        float v = acc[r] + bv;
        if (EPI == 1) v = (v > 20.f) ? v : log1pf(__expf(v));  // softplus
        if (EPI == 2) v = v * sc + sh;                         // fused BN
        C[(size_t)m * ldc + n] = v;
    }
}

// 2x2 register-blocked WMMA GEMM with software-pipelined fragment loads:
// one wave owns a 32x32 C macro-tile; next K-chunk's 8 b128 loads are issued
// before the current chunk's 4 v_wmma, so WMMAs never wait on in-flight loads.
template<int EPI>
__global__ __launch_bounds__(256) void k_gemm22(
    const _Float16* __restrict__ A, int lda,
    const _Float16* __restrict__ W, int ldw,
    float* __restrict__ C, int ldc,
    int Mt2, int Nt2, int Kloop, int Nbound,
    const float* __restrict__ bias,
    const float* __restrict__ scale,
    const float* __restrict__ shift)
{
    int wave = (int)((blockIdx.x * blockDim.x + threadIdx.x) >> 5);
    if (wave >= Mt2 * Nt2) return;             // wave-uniform exit (EXEC all-1s for WMMA)
    int lane = (int)(threadIdx.x & 31);
    int tm   = wave / Nt2;
    int tn   = wave % Nt2;
    int half = lane >> 4;
    int rc   = lane & 15;

    const _Float16* Ap0 = A + (size_t)(tm * 32 + rc) * lda + half * 8;
    const _Float16* Ap1 = Ap0 + (size_t)16 * lda;
    const _Float16* Wp0 = W + (size_t)(tn * 32 + rc) * ldw + half * 8;
    const _Float16* Wp1 = Wp0 + (size_t)16 * ldw;

    // prologue: chunk 0 fragments
    v8h a00 = *(const v8h*)(Ap0), a01 = *(const v8h*)(Ap0 + 16);
    v8h a10 = *(const v8h*)(Ap1), a11 = *(const v8h*)(Ap1 + 16);
    v8h b00 = *(const v8h*)(Wp0), b01 = *(const v8h*)(Wp0 + 16);
    v8h b10 = *(const v8h*)(Wp1), b11 = *(const v8h*)(Wp1 + 16);

    v8f acc00 = {}, acc01 = {}, acc10 = {}, acc11 = {};
    for (int kk = 0; kk < Kloop; ++kk) {
        // issue next chunk's loads first (clamped advance keeps tail in-bounds)
        int adv = (kk + 1 < Kloop) ? 32 : 0;
        Ap0 += adv; Ap1 += adv; Wp0 += adv; Wp1 += adv;
        v8h na00 = *(const v8h*)(Ap0), na01 = *(const v8h*)(Ap0 + 16);
        v8h na10 = *(const v8h*)(Ap1), na11 = *(const v8h*)(Ap1 + 16);
        v8h nb00 = *(const v8h*)(Wp0), nb01 = *(const v8h*)(Wp0 + 16);
        v8h nb10 = *(const v8h*)(Wp1), nb11 = *(const v8h*)(Wp1 + 16);

        // compute current chunk (no dependence on in-flight loads)
        v16h af0 = frag_combine(a00, a01);
        v16h af1 = frag_combine(a10, a11);
        v16h bf0 = frag_combine(b00, b01);
        v16h bf1 = frag_combine(b10, b11);
        acc00 = __builtin_amdgcn_wmma_f32_16x16x32_f16(false, af0, false, bf0,
                                                       (short)0, acc00, false, false);
        acc01 = __builtin_amdgcn_wmma_f32_16x16x32_f16(false, af0, false, bf1,
                                                       (short)0, acc01, false, false);
        acc10 = __builtin_amdgcn_wmma_f32_16x16x32_f16(false, af1, false, bf0,
                                                       (short)0, acc10, false, false);
        acc11 = __builtin_amdgcn_wmma_f32_16x16x32_f16(false, af1, false, bf1,
                                                       (short)0, acc11, false, false);

        a00 = na00; a01 = na01; a10 = na10; a11 = na11;
        b00 = nb00; b01 = nb01; b10 = nb10; b11 = nb11;
    }

    int n0 = tn * 32 + rc;
    int n1 = n0 + 16;
    epi_store<EPI>(C, ldc, tm * 32,      half, n0, Nbound, acc00, bias, scale, shift);
    epi_store<EPI>(C, ldc, tm * 32,      half, n1, Nbound, acc01, bias, scale, shift);
    epi_store<EPI>(C, ldc, tm * 32 + 16, half, n0, Nbound, acc10, bias, scale, shift);
    epi_store<EPI>(C, ldc, tm * 32 + 16, half, n1, Nbound, acc11, bias, scale, shift);
}

// 1x1 WMMA GEMM (for the tiny classifier GEMM with Mtiles == 1)
template<int EPI>
__global__ __launch_bounds__(256) void k_gemm(
    const _Float16* __restrict__ A, int lda,
    const _Float16* __restrict__ W, int ldw,
    float* __restrict__ C, int ldc,
    int Mtiles, int Ntiles, int Kloop, int Nbound,
    const float* __restrict__ bias,
    const float* __restrict__ scale,
    const float* __restrict__ shift)
{
    int wave = (int)((blockIdx.x * blockDim.x + threadIdx.x) >> 5);
    if (wave >= Mtiles * Ntiles) return;
    int lane = (int)(threadIdx.x & 31);
    int tm   = wave / Ntiles;
    int tn   = wave % Ntiles;
    int half = lane >> 4;
    int rc   = lane & 15;

    const _Float16* Ap = A + (size_t)(tm * 16 + rc) * lda + half * 8;
    const _Float16* Wp = W + (size_t)(tn * 16 + rc) * ldw + half * 8;

    v8f acc = {};
    for (int kk = 0; kk < Kloop; ++kk) {
        v8h a0 = *(const v8h*)(Ap);
        v8h a1 = *(const v8h*)(Ap + 16);
        v8h b0 = *(const v8h*)(Wp);
        v8h b1 = *(const v8h*)(Wp + 16);
        Ap += 32; Wp += 32;
        v16h af = frag_combine(a0, a1);
        v16h bf = frag_combine(b0, b1);
        acc = __builtin_amdgcn_wmma_f32_16x16x32_f16(false, af, false, bf,
                                                     (short)0, acc, false, false);
    }
    epi_store<EPI>(C, ldc, tm * 16, half, tn * 16 + rc, Nbound, acc, bias, scale, shift);
}

// x (B, IN, L) -> f16 (B*L, INP_) transposed + zero-padded
__global__ void k_xpose(const float* __restrict__ x, _Float16* __restrict__ xt)
{
    int e = (int)(blockIdx.x * blockDim.x + threadIdx.x);
    if (e >= ML_ * INP_) return;
    int c  = e % INP_;
    int bl = e / INP_;
    int b  = bl / L_;
    int t  = bl % L_;
    float v = (c < IN_) ? x[(size_t)(b * IN_ + c) * L_ + t] : 0.f;
    xt[e] = (_Float16)v;
}

// f32 (Nsrc,Ksrc) row-major -> f16 (Ndst,Kdst) zero-padded (also plain convert)
__global__ void k_padw(const float* __restrict__ src, _Float16* __restrict__ dst,
                       int Nsrc, int Ksrc, int Kdst, int total)
{
    int e = (int)(blockIdx.x * blockDim.x + threadIdx.x);
    if (e >= total) return;
    int n = e / Kdst, k = e % Kdst;
    dst[e] = (n < Nsrc && k < Ksrc) ? (_Float16)src[(size_t)n * Ksrc + k]
                                    : (_Float16)0.f;
}

__global__ void k_cvt16(const float* __restrict__ src, _Float16* __restrict__ dst, int n)
{
    int e = (int)(blockIdx.x * blockDim.x + threadIdx.x);
    if (e < n) dst[e] = (_Float16)src[e];
}

// depthwise causal conv (DC=4) + SiLU; dual f32/f16 store
__global__ void k_conv(const float* __restrict__ xc,
                       const float* __restrict__ cw, const float* __restrict__ cb,
                       float* __restrict__ xs, _Float16* __restrict__ xs16)
{
    int e = (int)(blockIdx.x * blockDim.x + threadIdx.x);
    if (e >= ML_ * DI_) return;
    int d    = e % DI_;
    int bl   = e / DI_;            // b*L + t
    int t    = bl % L_;
    int base = bl - t;             // b*L
    float acc = cb[d];
#pragma unroll
    for (int j = 0; j < DC_; ++j) {
        int tt = t - (DC_ - 1) + j;
        if (tt >= 0) acc += cw[d * DC_ + j] * xc[(size_t)(base + tt) * DI_ + d];
    }
    float v = acc / (1.f + __expf(-acc));      // SiLU
    xs[e]   = v;
    xs16[e] = (_Float16)v;
}

// BN folded into per-channel scale/shift
__global__ void k_bnprep(const float* __restrict__ g,  const float* __restrict__ be,
                         const float* __restrict__ mu, const float* __restrict__ var,
                         float* __restrict__ scale, float* __restrict__ shift)
{
    int i = (int)(blockIdx.x * blockDim.x + threadIdx.x);
    if (i >= DM_) return;
    float s  = g[i] * rsqrtf(var[i] + 1e-5f);
    scale[i] = s;
    shift[i] = be[i] - mu[i] * s;
}

// Selective scan: thread = one (b, d) channel, DS=16 state in registers.
// B_t/C_t broadcast through LDS. Fuses D skip + SiLU(z) gate, stores f16.
__global__ __launch_bounds__(256) void k_scan(
    const float* __restrict__ dt,  const float* __restrict__ xs,
    const float* __restrict__ z,   const float* __restrict__ dbc,
    const float* __restrict__ A_log, const float* __restrict__ Dp,
    _Float16* __restrict__ y16)
{
    __shared__ float sB[DS_], sC[DS_];
    int b     = (int)(blockIdx.x / (DI_ / 256));
    int chunk = (int)(blockIdx.x % (DI_ / 256));
    int d     = chunk * 256 + (int)threadIdx.x;

    float Ai[DS_], s[DS_];
#pragma unroll
    for (int i = 0; i < DS_; ++i) {
        Ai[i] = -__expf(A_log[(size_t)d * DS_ + i]);
        s[i]  = 0.f;
    }
    float dval = Dp[d];

    for (int t = 0; t < L_; ++t) {
        int row = b * L_ + t;
        if (threadIdx.x < DS_)
            sB[threadIdx.x] = dbc[(size_t)row * 64 + DTR_ + threadIdx.x];
        else if (threadIdx.x < 2 * DS_)
            sC[threadIdx.x - DS_] = dbc[(size_t)row * 64 + DTR_ + DS_ + (threadIdx.x - DS_)];
        __syncthreads();

        size_t idx = (size_t)row * DI_ + d;
        float dtv = dt[idx];
        float xv  = xs[idx];
        float zv  = z[idx];
        float dx  = dtv * xv;
        float y   = 0.f;
#pragma unroll
        for (int i = 0; i < DS_; ++i) {
            s[i] = __expf(dtv * Ai[i]) * s[i] + dx * sB[i];
            y   += s[i] * sC[i];
        }
        float out = (y + xv * dval) * (zv / (1.f + __expf(-zv)));
        y16[idx]  = (_Float16)out;
        __syncthreads();
    }
}

// mean over L -> pooled f16 (B x DM)
__global__ void k_pool(const float* __restrict__ h, _Float16* __restrict__ pooled16)
{
    int e = (int)(blockIdx.x * blockDim.x + threadIdx.x);
    if (e >= B_ * DM_) return;
    int b = e / DM_, dm = e % DM_;
    float sum = 0.f;
    for (int t = 0; t < L_; ++t) sum += h[(size_t)(b * L_ + t) * DM_ + dm];
    pooled16[e] = (_Float16)(sum * (1.f / L_));
}

// ---------------------------------------------------------------------------
extern "C" void kernel_launch(void* const* d_in, const int* in_sizes, int n_in,
                              void* d_out, int out_size, void* d_ws, size_t ws_size,
                              hipStream_t stream)
{
    (void)in_sizes; (void)n_in; (void)out_size; (void)ws_size;
    const float* x        = (const float*)d_in[0];
    const float* proj_W   = (const float*)d_in[1];
    const float* proj_b   = (const float*)d_in[2];
    const float* in_projW = (const float*)d_in[3];
    const float* conv_w   = (const float*)d_in[4];
    const float* conv_b   = (const float*)d_in[5];
    const float* x_projW  = (const float*)d_in[6];
    const float* dt_projW = (const float*)d_in[7];
    const float* dt_projb = (const float*)d_in[8];
    const float* A_log    = (const float*)d_in[9];
    const float* Dvec     = (const float*)d_in[10];
    const float* out_projW= (const float*)d_in[11];
    const float* bn_g     = (const float*)d_in[12];
    const float* bn_b     = (const float*)d_in[13];
    const float* bn_m     = (const float*)d_in[14];
    const float* bn_v     = (const float*)d_in[15];
    const float* fc_W     = (const float*)d_in[16];
    const float* fc_b     = (const float*)d_in[17];
    float* out = (float*)d_out;

    size_t off = 0;
    auto alloc = [&](size_t bytes) -> void* {
        void* p = (char*)d_ws + off;
        off += (bytes + 255) & ~(size_t)255;
        return p;
    };
    _Float16* xt16     = (_Float16*)alloc((size_t)ML_ * INP_ * 2);
    _Float16* pW16     = (_Float16*)alloc((size_t)DM_ * INP_ * 2);
    _Float16* Wi16     = (_Float16*)alloc((size_t)2 * DI_ * DM_ * 2);
    _Float16* Wx16     = (_Float16*)alloc((size_t)(DTR_ + 2 * DS_) * DI_ * 2);
    _Float16* Wdt16    = (_Float16*)alloc((size_t)DI_ * DTR_ * 2);
    _Float16* Wo16     = (_Float16*)alloc((size_t)DM_ * DI_ * 2);
    _Float16* Wfc16    = (_Float16*)alloc((size_t)NFCP_ * DM_ * 2);
    float*    bnscale  = (float*)alloc((size_t)DM_ * 4);
    float*    bnshift  = (float*)alloc((size_t)DM_ * 4);
    float*    hbuf     = (float*)alloc((size_t)ML_ * DM_ * 4);
    _Float16* h16      = (_Float16*)alloc((size_t)ML_ * DM_ * 2);
    float*    xcbuf    = (float*)alloc((size_t)ML_ * DI_ * 4);   // also reused as dt
    float*    zbuf     = (float*)alloc((size_t)ML_ * DI_ * 4);
    float*    xsbuf    = (float*)alloc((size_t)ML_ * DI_ * 4);
    _Float16* xs16     = (_Float16*)alloc((size_t)ML_ * DI_ * 2);
    float*    dbc      = (float*)alloc((size_t)ML_ * 64 * 4);
    _Float16* dbc16    = (_Float16*)alloc((size_t)ML_ * 64 * 2);
    _Float16* y16      = (_Float16*)alloc((size_t)ML_ * DI_ * 2);
    _Float16* pooled16 = (_Float16*)alloc((size_t)B_ * DM_ * 2);

    auto nb = [](int n) { return dim3((unsigned)((n + 255) / 256)); };
    auto gb = [](int waves) { return dim3((unsigned)((waves + 7) / 8)); }; // 8 waves/block
    const dim3 T(256);
    const int Mt2 = ML_ / 32;   // 256 macro-tile rows of 32

    // --- input projection: h = x^T @ proj_W^T + proj_b ---
    k_xpose<<<nb(ML_ * INP_), T, 0, stream>>>(x, xt16);
    k_padw<<<nb(DM_ * INP_), T, 0, stream>>>(proj_W, pW16, DM_, IN_, INP_, DM_ * INP_);
    k_gemm22<0><<<gb(Mt2 * (DM_ / 32)), T, 0, stream>>>(
        xt16, INP_, pW16, INP_, hbuf, DM_,
        Mt2, DM_ / 32, INP_ / 32, DM_, proj_b, nullptr, nullptr);

    for (int l = 0; l < NL_; ++l) {
        // per-layer weight conversion (trivial vs. GEMM cost; keeps ws small)
        k_padw<<<nb(2 * DI_ * DM_), T, 0, stream>>>(
            in_projW + (size_t)l * 2 * DI_ * DM_, Wi16, 2 * DI_, DM_, DM_, 2 * DI_ * DM_);
        k_padw<<<nb((DTR_ + 2 * DS_) * DI_), T, 0, stream>>>(
            x_projW + (size_t)l * (DTR_ + 2 * DS_) * DI_, Wx16,
            DTR_ + 2 * DS_, DI_, DI_, (DTR_ + 2 * DS_) * DI_);
        k_padw<<<nb(DI_ * DTR_), T, 0, stream>>>(
            dt_projW + (size_t)l * DI_ * DTR_, Wdt16, DI_, DTR_, DTR_, DI_ * DTR_);
        k_padw<<<nb(DM_ * DI_), T, 0, stream>>>(
            out_projW + (size_t)l * DM_ * DI_, Wo16, DM_, DI_, DI_, DM_ * DI_);

        // h -> f16
        k_cvt16<<<nb(ML_ * DM_), T, 0, stream>>>(hbuf, h16, ML_ * DM_);

        // in_proj: xc and z (M=8192, K=512, N=1024 each)
        k_gemm22<0><<<gb(Mt2 * (DI_ / 32)), T, 0, stream>>>(
            h16, DM_, Wi16, DM_, xcbuf, DI_,
            Mt2, DI_ / 32, DM_ / 32, DI_, nullptr, nullptr, nullptr);
        k_gemm22<0><<<gb(Mt2 * (DI_ / 32)), T, 0, stream>>>(
            h16, DM_, Wi16 + (size_t)DI_ * DM_, DM_, zbuf, DI_,
            Mt2, DI_ / 32, DM_ / 32, DI_, nullptr, nullptr, nullptr);

        // depthwise causal conv + SiLU
        k_conv<<<nb(ML_ * DI_), T, 0, stream>>>(
            xcbuf, conv_w + (size_t)l * DI_ * DC_, conv_b + (size_t)l * DI_, xsbuf, xs16);

        // x_proj: dbc (M=8192, K=1024, N=64)
        k_gemm22<0><<<gb(Mt2 * 2), T, 0, stream>>>(
            xs16, DI_, Wx16, DI_, dbc, 64,
            Mt2, 2, DI_ / 32, 64, nullptr, nullptr, nullptr);
        k_cvt16<<<nb(ML_ * 64), T, 0, stream>>>(dbc, dbc16, ML_ * 64);

        // dt_proj + softplus (M=8192, K=32 of lda=64, N=1024) -> reuse xcbuf
        k_gemm22<1><<<gb(Mt2 * (DI_ / 32)), T, 0, stream>>>(
            dbc16, 64, Wdt16, DTR_, xcbuf, DI_,
            Mt2, DI_ / 32, 1, DI_, dt_projb + (size_t)l * DI_, nullptr, nullptr);

        // selective scan, fused D-skip and SiLU(z) gate
        k_scan<<<dim3(B_ * (DI_ / 256)), T, 0, stream>>>(
            xcbuf, xsbuf, zbuf, dbc,
            A_log + (size_t)l * DI_ * DS_, Dvec + (size_t)l * DI_, y16);

        // out_proj with fused BN epilogue (M=8192, K=1024, N=512) -> hbuf
        k_bnprep<<<dim3(2), T, 0, stream>>>(
            bn_g + (size_t)l * DM_, bn_b + (size_t)l * DM_,
            bn_m + (size_t)l * DM_, bn_v + (size_t)l * DM_, bnscale, bnshift);
        k_gemm22<2><<<gb(Mt2 * (DM_ / 32)), T, 0, stream>>>(
            y16, DI_, Wo16, DI_, hbuf, DM_,
            Mt2, DM_ / 32, DI_ / 32, DM_, nullptr, bnscale, bnshift);
    }

    // mean pool + classifier (M=16, K=512, N=35 padded to 48) -> 1x1 kernel
    k_pool<<<nb(B_ * DM_), T, 0, stream>>>(hbuf, pooled16);
    k_padw<<<nb(NFCP_ * DM_), T, 0, stream>>>(fc_W, Wfc16, NLAB_, DM_, DM_, NFCP_ * DM_);
    k_gemm<0><<<gb(NFCP_ / 16), T, 0, stream>>>(
        pooled16, DM_, Wfc16, DM_, out, NLAB_,
        1, NFCP_ / 16, DM_ / 32, NLAB_, fc_b, nullptr, nullptr);
}
